// SpatialGatingUnit_91027536871595
// MI455X (gfx1250) — compile-verified, matrix operand order checked by reading, not run on hardware
//
#include <hip/hip_runtime.h>
#include <hip/hip_bf16.h>

// SpatialGatingUnit for MI455X (gfx1250): bf16 WMMA GEMMs + f32 VALU epilogues.
// 512 images, one workgroup (8 wave32) per image.
// v2: 2x4 wave tiling (halved redundant B traffic), software-pipelined B frags,
//     global_prefetch priming of weight strips.

typedef __attribute__((ext_vector_type(16))) __bf16        v16bf;
typedef __attribute__((ext_vector_type(8)))  float         v8f;
typedef __attribute__((ext_vector_type(4)))  unsigned int  u32x4;

// ---- workspace layout (bytes) ----
#define OFF_C0   0          // conv0_w bf16 [256][256]
#define OFF_C2   131072     // conv2_w bf16 [256][256]
#define OFF_W1   262144     // w1_w * bn_scale, bf16 [64][256]
#define OFF_W2   294912     // w2_w padded,     bf16 [64][64] (rows 49..63 = 0)
#define OFF_B1   303104     // folded bias,     f32  [64]
#define WS_BYTES 303360

// ---- LDS layout (bytes); regions alias across lifetimes ----
#define SA_OFF   0          // __bf16 [64][264]  (LN-gate -> g1 bf16 -> dw-out bf16)
#define ST_OFF   33792      // __bf16 [64][72]   (relu(t) bf16)
#define SWD_OFF  43008      // float  [64][56]   (wdyn, 49 taps)
#define SGF_OFF  65792      // float  [64][264]  (gate f32 -> g1 f32)
#define SMEM_BYTES 133376
// sXdwT (float [64][257], 65792B) aliases [0, SGF_OFF), dead after stage A1.

union FragU { v16bf v; u32x4 q[2]; };

// Load a 16x32 bf16 WMMA operand fragment.
// ISA 16-bit A layout: lane&15 = row, lane>>4 = half; per lane the 16 K values
// are {kbase+half*8 .. +7} then {kbase+16+half*8 .. +7}, 2 per VGPR -> two b128s.
__device__ __forceinline__ v16bf load_frag(const unsigned char* base,
                                           int rowStrideBytes, int kbase, int lane) {
  const int r = lane & 15, half = lane >> 4;
  const unsigned char* rp = base + r * rowStrideBytes;
  FragU f;
  f.q[0] = *(const u32x4*)(rp + (kbase +      half * 8) * 2);
  f.q[1] = *(const u32x4*)(rp + (kbase + 16 + half * 8) * 2);
  return f.v;
}

__device__ __forceinline__ v8f wmma_bf16(v16bf a, v16bf b, v8f c) {
  return __builtin_amdgcn_wmma_f32_16x16x32_bf16(false, a, false, b, (short)0, c,
                                                 false, false);
}

// 64x256x256 GEMM strip for one wave: rows mg*32..+31, cols ng*64..+63.
// A from LDS (row stride 528B), B from global weights (row stride 512B).
// One-deep B-fragment lookahead so each load overlaps at least one WMMA.
__device__ __forceinline__ void gemm_2x4(const __bf16* sAp, int mg, int ng,
                                         int lane, const unsigned char* Bw,
                                         v8f (&acc)[2][4]) {
  const unsigned char* A0 = (const unsigned char*)sAp + mg * 32 * 528;
  const unsigned char* A1 = A0 + 16 * 528;
  const unsigned char* Bb = Bw + ng * 64 * 512;
  v16bf a0 = load_frag(A0, 528, 0, lane);
  v16bf a1 = load_frag(A1, 528, 0, lane);
  v16bf b  = load_frag(Bb, 512, 0, lane);
#pragma unroll
  for (int kk = 0; kk < 8; ++kk) {
    const int kb = kk * 32;
#pragma unroll
    for (int s = 0; s < 4; ++s) {
      v16bf bn = b;
      if (s < 3)       bn = load_frag(Bb + (s + 1) * 16 * 512, 512, kb, lane);
      else if (kk < 7) bn = load_frag(Bb, 512, kb + 32, lane);
      acc[0][s] = wmma_bf16(a0, b, acc[0][s]);
      acc[1][s] = wmma_bf16(a1, b, acc[1][s]);
      b = bn;
    }
    if (kk < 7) {
      a0 = load_frag(A0, 528, kb + 32, lane);
      a1 = load_frag(A1, 528, kb + 32, lane);
    }
  }
}

// ---------------- weight prep: f32 -> bf16, fold BN into w1 ----------------
__global__ void prep_weights(const float* __restrict__ conv0_w,
                             const float* __restrict__ conv2_w,
                             const float* __restrict__ w1_w,
                             const float* __restrict__ w1_b,
                             const float* __restrict__ bn_g,
                             const float* __restrict__ bn_b,
                             const float* __restrict__ bn_m,
                             const float* __restrict__ bn_v,
                             const float* __restrict__ w2_w,
                             unsigned char* __restrict__ ws) {
  const int idx = blockIdx.x * blockDim.x + threadIdx.x;   // 0..65535
  __bf16* c0 = (__bf16*)(ws + OFF_C0);
  __bf16* c2 = (__bf16*)(ws + OFF_C2);
  __bf16* w1 = (__bf16*)(ws + OFF_W1);
  __bf16* w2 = (__bf16*)(ws + OFF_W2);
  float*  b1 = (float*) (ws + OFF_B1);
  if (idx < 65536) {
    c0[idx] = (__bf16)conv0_w[idx];
    c2[idx] = (__bf16)conv2_w[idx];
  }
  if (idx < 16384) {
    const int r = idx >> 8;
    const float s = bn_g[r] * rsqrtf(bn_v[r] + 1e-5f);
    w1[idx] = (__bf16)(w1_w[idx] * s);
  }
  if (idx < 4096) {
    const int j = idx >> 6, rr = idx & 63;
    w2[idx] = (j < 49) ? (__bf16)w2_w[j * 64 + rr] : (__bf16)0.0f;
  }
  if (idx < 64) {
    const float s = bn_g[idx] * rsqrtf(bn_v[idx] + 1e-5f);
    b1[idx] = (w1_b[idx] - bn_m[idx]) * s + bn_b[idx];
  }
}

// ---------------- main fused kernel: one block per image ----------------
__global__ __launch_bounds__(256, 1)
void sgu_main(const float* __restrict__ x,        // (512, 64, 512)
              const float* __restrict__ xdw_g,    // (512, 256, 64)
              const float* __restrict__ ln_g,
              const float* __restrict__ ln_b,
              const float* __restrict__ w2_b,     // (49,)
              const unsigned char* __restrict__ ws,
              float* __restrict__ out) {          // (512, 64, 256)
  extern __shared__ unsigned char smem[];
  __bf16* sA    = (__bf16*)(smem + SA_OFF);
  __bf16* sT    = (__bf16*)(smem + ST_OFF);
  float*  sWd   = (float*) (smem + SWD_OFF);
  float*  sGf   = (float*) (smem + SGF_OFF);   // also g1 f32 after conv0
  float*  sXdwT = (float*) (smem);             // stride 257, dead after A1

  const int b    = blockIdx.x;
  const int t    = threadIdx.x;
  const int lane = t & 31;
  const int wv   = t >> 5;

  const float* xb   = x     + (size_t)b * 32768;   // row p at +p*512
  const float* xdwb = xdw_g + (size_t)b * 16384;   // [c][p]

  // prime WGP$ with conv0 + w1 weights (consumed right after LN)
  {
    const unsigned char* p0 = ws + OFF_C0 + t * 512;  // 256 threads x 512B = 128KB
#pragma unroll
    for (int i = 0; i < 8; ++i) __builtin_prefetch(p0 + i * 64, 0, 1);
    const unsigned char* p1 = ws + OFF_W1 + t * 128;  // 32KB
#pragma unroll
    for (int i = 0; i < 2; ++i) __builtin_prefetch(p1 + i * 64, 0, 1);
  }

  // ---- A0: stage xdw transposed into LDS (coalesced global read) ----
  for (int i = 0; i < 64; ++i) {
    const int idx = t + (i << 8);
    const int c = idx >> 6, p = idx & 63;
    sXdwT[p * 257 + c] = xdwb[idx];
  }
  __syncthreads();

  // ---- A1: gate = x[...,256:] * xdw (coalesced x read) ----
  for (int i = 0; i < 64; ++i) {
    const int idx = t + (i << 8);
    const int p = idx >> 8, c = idx & 255;
    sGf[p * 264 + c] = xb[p * 512 + 256 + c] * sXdwT[p * 257 + c];
  }
  __syncthreads();

  // ---- A2: LayerNorm over channels; write bf16 A operand ----
  {
    const int p = t >> 2, q = t & 3;
    const float* row = &sGf[p * 264 + q * 64];
    float s = 0.f, ss = 0.f;
#pragma unroll 8
    for (int j = 0; j < 64; ++j) { const float v = row[j]; s += v; ss += v * v; }
    s  += __shfl_xor(s, 1, 32);   s  += __shfl_xor(s, 2, 32);
    ss += __shfl_xor(ss, 1, 32);  ss += __shfl_xor(ss, 2, 32);
    const float mu  = s * (1.f / 256.f);
    const float var = ss * (1.f / 256.f) - mu * mu;
    const float rs  = rsqrtf(var + 1e-5f);
#pragma unroll 4
    for (int j = 0; j < 64; ++j) {
      const int c = q * 64 + j;
      sA[p * 264 + c] = (__bf16)((row[j] - mu) * rs * ln_g[c] + ln_b[c]);
    }
  }
  __syncthreads();

  const int mg = wv & 1;        // M half: rows mg*32..+31
  const int ng = wv >> 1;       // N group: cols ng*64..+63 (256-wide GEMMs)

  // ---- conv0: g1[p][o] = sum_c gate[p][c] * conv0[o][c] ----
  v8f acc[2][4];
#pragma unroll
  for (int u = 0; u < 2; ++u)
#pragma unroll
    for (int s = 0; s < 4; ++s) acc[u][s] = (v8f){0, 0, 0, 0, 0, 0, 0, 0};
  gemm_2x4(sA, mg, ng, lane, ws + OFF_C0, acc);
  __syncthreads();              // all reads of sA (LN-gate) done
  // write g1: f32 (for dwconv) + bf16 (A operand of w1 GEMM)
#pragma unroll
  for (int u = 0; u < 2; ++u)
#pragma unroll
    for (int s = 0; s < 4; ++s) {
      const int n = ng * 64 + s * 16 + (lane & 15);
#pragma unroll
      for (int j = 0; j < 8; ++j) {
        const int m = mg * 32 + u * 16 + ((lane >> 4) << 3) + j;
        const float v = acc[u][s][j];
        sGf[m * 264 + n] = v;
        sA [m * 264 + n] = (__bf16)v;
      }
    }
  __syncthreads();

  // ---- w1 GEMM + bias + ReLU: t[p][r] (M=64,N=64,K=256) ----
  {
    const int mi4 = wv & 3;
    const int nb  = (wv >> 2) * 32;
    v8f a2[2];
    a2[0] = (v8f){0, 0, 0, 0, 0, 0, 0, 0};
    a2[1] = (v8f){0, 0, 0, 0, 0, 0, 0, 0};
    const unsigned char* Ac = (const unsigned char*)sA + mi4 * 16 * 528;
    const unsigned char* B1 = ws + OFF_W1;
#pragma unroll
    for (int kk = 0; kk < 8; ++kk) {
      const int kbase = kk * 32;
      const v16bf af = load_frag(Ac, 528, kbase, lane);
#pragma unroll
      for (int s = 0; s < 2; ++s) {
        const v16bf bf = load_frag(B1 + (nb + s * 16) * 512, 512, kbase, lane);
        a2[s] = wmma_bf16(af, bf, a2[s]);
      }
    }
    const float* b1 = (const float*)(ws + OFF_B1);
#pragma unroll
    for (int s = 0; s < 2; ++s) {
      const int r = nb + s * 16 + (lane & 15);
      const float bb = b1[r];
#pragma unroll
      for (int j = 0; j < 8; ++j) {
        const int m = mi4 * 16 + ((lane >> 4) << 3) + j;
        float v = a2[s][j] + bb;
        sT[m * 72 + r] = (__bf16)(v > 0.f ? v : 0.f);
      }
    }
  }
  __syncthreads();

  // ---- w2 GEMM + bias: wdyn[p][tap] (M=64,N=49(pad 64),K=64) ----
  {
    const int mi4 = wv & 3;
    const int nb  = (wv >> 2) * 32;
    v8f a2[2];
    a2[0] = (v8f){0, 0, 0, 0, 0, 0, 0, 0};
    a2[1] = (v8f){0, 0, 0, 0, 0, 0, 0, 0};
    const unsigned char* At = (const unsigned char*)sT + mi4 * 16 * 144;
    const unsigned char* B2 = ws + OFF_W2;
#pragma unroll
    for (int kk = 0; kk < 2; ++kk) {
      const int kbase = kk * 32;
      const v16bf af = load_frag(At, 144, kbase, lane);
#pragma unroll
      for (int s = 0; s < 2; ++s) {
        const v16bf bf = load_frag(B2 + (nb + s * 16) * 128, 128, kbase, lane);
        a2[s] = wmma_bf16(af, bf, a2[s]);
      }
    }
#pragma unroll
    for (int s = 0; s < 2; ++s) {
      const int n = nb + s * 16 + (lane & 15);
      if (n < 49) {
        const float wb = w2_b[n];
#pragma unroll
        for (int j = 0; j < 8; ++j) {
          const int m = mi4 * 16 + ((lane >> 4) << 3) + j;
          sWd[m * 56 + n] = a2[s][j] + wb;
        }
      }
    }
  }
  __syncthreads();

  // stream conv2 weights toward WGP$ while the VALU-heavy dwconv runs
  {
    const unsigned char* p2 = ws + OFF_C2 + t * 512;  // 128KB
#pragma unroll
    for (int i = 0; i < 8; ++i) __builtin_prefetch(p2 + i * 64, 0, 1);
  }

  // ---- dynamic 7x7 depthwise conv (f32 VALU): one channel per thread ----
  {
    const int c = t;                         // 0..255
    for (int p = 0; p < 64; ++p) {
      const int h = p >> 3, w = p & 7;
      const float* wrow = &sWd[p * 56];
      float av = 0.f;
#pragma unroll
      for (int ki = 0; ki < 7; ++ki) {
        const int hp = h + ki - 3;
        if ((unsigned)hp < 8u) {
#pragma unroll
          for (int kj = 0; kj < 7; ++kj) {
            const int wp = w + kj - 3;
            if ((unsigned)wp < 8u)
              av += wrow[ki * 7 + kj] * sGf[(hp * 8 + wp) * 264 + c];
          }
        }
      }
      sA[p * 264 + c] = (__bf16)av;          // A operand for conv2
    }
  }
  __syncthreads();

  // ---- conv2 GEMM + residual-multiply epilogue ----
#pragma unroll
  for (int u = 0; u < 2; ++u)
#pragma unroll
    for (int s = 0; s < 4; ++s) acc[u][s] = (v8f){0, 0, 0, 0, 0, 0, 0, 0};
  gemm_2x4(sA, mg, ng, lane, ws + OFF_C2, acc);
  {
    float* ob = out + (size_t)b * 16384;
#pragma unroll
    for (int u = 0; u < 2; ++u)
#pragma unroll
      for (int s = 0; s < 4; ++s) {
        const int n = ng * 64 + s * 16 + (lane & 15);
#pragma unroll
        for (int j = 0; j < 8; ++j) {
          const int m = mg * 32 + u * 16 + ((lane >> 4) << 3) + j;
          ob[m * 256 + n] = acc[u][s][j] * xb[m * 512 + n];  // res = x[..., :256]
        }
      }
  }
}

extern "C" void kernel_launch(void* const* d_in, const int* in_sizes, int n_in,
                              void* d_out, int out_size, void* d_ws, size_t ws_size,
                              hipStream_t stream) {
  (void)in_sizes; (void)n_in; (void)out_size; (void)ws_size;
  const float* x    = (const float*)d_in[0];
  const float* xdw  = (const float*)d_in[1];
  const float* lng  = (const float*)d_in[2];
  const float* lnb  = (const float*)d_in[3];
  const float* c0w  = (const float*)d_in[4];
  const float* c2w  = (const float*)d_in[5];
  const float* w1w  = (const float*)d_in[6];
  const float* w1b  = (const float*)d_in[7];
  const float* bng  = (const float*)d_in[8];
  const float* bnb  = (const float*)d_in[9];
  const float* bnm  = (const float*)d_in[10];
  const float* bnv  = (const float*)d_in[11];
  const float* w2w  = (const float*)d_in[12];
  const float* w2b  = (const float*)d_in[13];
  unsigned char* ws = (unsigned char*)d_ws;
  float* out        = (float*)d_out;

  prep_weights<<<256, 256, 0, stream>>>(c0w, c2w, w1w, w1b, bng, bnb, bnm, bnv,
                                        w2w, ws);
  sgu_main<<<512, 256, SMEM_BYTES, stream>>>(x, xdw, lng, lnb, w2b, ws, out);
}
